// SageNet_54056458387938
// MI455X (gfx1250) — matrix operand order repaired and checked by reading, not compiled, not run on hardware
//
#include <hip/hip_runtime.h>

// ---------------------------------------------------------------------------
// Problem constants (from reference)
// ---------------------------------------------------------------------------
#define N_SRC0 200000
#define N_DST0 40000
#define N_SRC1 40000
#define N_DST1 8000
#define E0     1000000
#define E1     80000
#define D_IN   256
#define D_HID  512
#define D_OUT  256

typedef __attribute__((ext_vector_type(16))) __bf16 v16bf;
typedef __attribute__((ext_vector_type(8)))  float  v8f;

// ---------------------------------------------------------------------------
// bf16 helpers (round-to-nearest-even), stored as unsigned short
// ---------------------------------------------------------------------------
__device__ __forceinline__ unsigned short f2bf(float f) {
    unsigned int u = __float_as_uint(f);
    u += 0x7FFFu + ((u >> 16) & 1u);
    return (unsigned short)(u >> 16);
}
__device__ __forceinline__ float bf2f(unsigned short h) {
    return __uint_as_float(((unsigned int)h) << 16);
}

// ---------------------------------------------------------------------------
// Weight prep (pre-TRANSPOSED so GEMM B-tiles are contiguous row loads):
//   W0t[n][k] (512x512 bf16)  : k<256 -> Wself0[k][n], else Wneigh0[k-256][n]
//   W1t[n][k] (256x1024 bf16) : k<512 -> Wself1[k][n], else Wneigh1[k-512][n]
// ---------------------------------------------------------------------------
__global__ __launch_bounds__(256) void cvt_w0_kernel(
    const float* __restrict__ Wself, const float* __restrict__ Wneigh,
    unsigned short* __restrict__ Wt)
{
    int i = blockIdx.x * 256 + threadIdx.x;           // over 512*512 (n,k)
    if (i >= 512 * 512) return;
    int n = i >> 9, k = i & 511;
    float v = (k < 256) ? Wself[k * 512 + n] : Wneigh[(k - 256) * 512 + n];
    Wt[i] = f2bf(v);
}

__global__ __launch_bounds__(256) void cvt_w1_kernel(
    const float* __restrict__ Wself, const float* __restrict__ Wneigh,
    unsigned short* __restrict__ Wt)
{
    int i = blockIdx.x * 256 + threadIdx.x;           // over 256*1024 (n,k)
    if (i >= 256 * 1024) return;
    int n = i >> 10, k = i & 1023;
    float v = (k < 512) ? Wself[k * 256 + n] : Wneigh[(k - 512) * 256 + n];
    Wt[i] = f2bf(v);
}

// ---------------------------------------------------------------------------
// Layer-0 scatter: one wave32 per edge, 256 f32 feats -> 8 per lane.
// msg0[dst] += feat[src]; deg0[dst] += 1   (no-return global_atomic_add_f32)
// ---------------------------------------------------------------------------
__global__ __launch_bounds__(256) void scatter0_kernel(
    const float* __restrict__ feat, const int* __restrict__ esrc,
    const int* __restrict__ edst, float* __restrict__ msg,
    float* __restrict__ deg, int E)
{
    int e    = (blockIdx.x * 256 + threadIdx.x) >> 5;
    int lane = threadIdx.x & 31;
    if (e >= E) return;
    int s = esrc[e], d = edst[e];
    const float4* fs = (const float4*)(feat + (size_t)s * D_IN) + lane * 2;
    float* md = msg + (size_t)d * D_IN + lane * 8;
    __builtin_prefetch(md, 0, 0);                     // global_prefetch_b8
    float4 v0 = fs[0];
    float4 v1 = fs[1];
    atomicAdd(md + 0, v0.x); atomicAdd(md + 1, v0.y);
    atomicAdd(md + 2, v0.z); atomicAdd(md + 3, v0.w);
    atomicAdd(md + 4, v1.x); atomicAdd(md + 5, v1.y);
    atomicAdd(md + 6, v1.z); atomicAdd(md + 7, v1.w);
    if (lane == 0) atomicAdd(deg + d, 1.0f);
}

// ---------------------------------------------------------------------------
// Layer-1 scatter: one wave32 per edge, 512 bf16 feats -> 16 per lane.
// ---------------------------------------------------------------------------
__global__ __launch_bounds__(256) void scatter1_kernel(
    const unsigned short* __restrict__ H, const int* __restrict__ esrc,
    const int* __restrict__ edst, float* __restrict__ msg,
    float* __restrict__ deg, int E)
{
    int e    = (blockIdx.x * 256 + threadIdx.x) >> 5;
    int lane = threadIdx.x & 31;
    if (e >= E) return;
    int s = esrc[e], d = edst[e];
    const uint4* hs = (const uint4*)(H + (size_t)s * D_HID + lane * 16);
    float* md = msg + (size_t)d * D_HID + lane * 16;
    uint4 qa = hs[0];
    uint4 qb = hs[1];
    unsigned int w[8] = {qa.x, qa.y, qa.z, qa.w, qb.x, qb.y, qb.z, qb.w};
#pragma unroll
    for (int i = 0; i < 8; ++i) {
        atomicAdd(md + 2 * i + 0, bf2f((unsigned short)(w[i] & 0xFFFFu)));
        atomicAdd(md + 2 * i + 1, bf2f((unsigned short)(w[i] >> 16)));
    }
    if (lane == 0) atomicAdd(deg + d, 1.0f);
}

// ---------------------------------------------------------------------------
// X0[n, 0:256]   = bf16(feat[n]);  X0[n, 256:512] = bf16(msg0[n]/max(deg,1))
// ---------------------------------------------------------------------------
__global__ __launch_bounds__(256) void build_x0_kernel(
    const float* __restrict__ feat, const float* __restrict__ msg,
    const float* __restrict__ deg, unsigned short* __restrict__ X0)
{
    size_t i = (size_t)blockIdx.x * 256 + threadIdx.x;   // over 40000*256
    if (i >= (size_t)N_DST0 * D_IN) return;
    int n = (int)(i >> 8), c = (int)(i & 255);
    float r = 1.0f / fmaxf(deg[n], 1.0f);
    X0[(size_t)n * (2 * D_IN) + c]        = f2bf(feat[i]);
    X0[(size_t)n * (2 * D_IN) + D_IN + c] = f2bf(msg[i] * r);
}

// ---------------------------------------------------------------------------
// X1[n, 0:512] = H[n];  X1[n, 512:1024] = bf16(msg1[n]/max(deg,1))
// ---------------------------------------------------------------------------
__global__ __launch_bounds__(256) void build_x1_kernel(
    const unsigned short* __restrict__ H, const float* __restrict__ msg,
    const float* __restrict__ deg, unsigned short* __restrict__ X1)
{
    size_t i = (size_t)blockIdx.x * 256 + threadIdx.x;   // over 8000*512
    if (i >= (size_t)N_DST1 * D_HID) return;
    int n = (int)(i >> 9), c = (int)(i & 511);
    float r = 1.0f / fmaxf(deg[n], 1.0f);
    X1[(size_t)n * (2 * D_HID) + c]         = H[(size_t)n * D_HID + c];
    X1[(size_t)n * (2 * D_HID) + D_HID + c] = f2bf(msg[i] * r);
}

// ---------------------------------------------------------------------------
// bf16 WMMA GEMM:  C(MxN) = A(MxK,bf16) @ Bt(NxK,bf16)^T + bias, opt. ReLU.
// Block tile 128x64, 8 waves (4 M x 2 N), each wave 32x32 = 2x2 WMMA tiles.
// Double-buffered GLOBAL_LOAD_ASYNC_TO_LDS_B128 staging (ASYNCcnt tracked):
// 3 async ops per wave per K-step; in-order completion => wait asynccnt<=3
// guarantees the current tile landed while the prefetch is in flight.
// ---------------------------------------------------------------------------
#define KPAD 40   // 32 + 8 bf16 padding: row stride 80B, conflict-free, 16B-aligned

union Frag {
    v16bf v;
    uint4 q[2];
};

// LDS byte address = low 32 bits of the generic (flat) shared-memory address
// (flat aperture rule).
__device__ __forceinline__ unsigned lds_addr32(const void* p) {
    return (unsigned)(unsigned long long)p;
}

__device__ __forceinline__ void async_cp16(unsigned lds_off, const void* gaddr) {
    asm volatile("global_load_async_to_lds_b128 %0, %1, off"
                 :: "v"(lds_off), "v"(gaddr) : "memory");
}

__device__ __forceinline__ void stage_tile_async(
    const unsigned short* __restrict__ A, const unsigned short* __restrict__ Bt,
    unsigned short* AsBuf, unsigned short* BsBuf,
    int blockM, int blockN, int M, int K, int k0, int t)
{
    // A tile: 128 rows x 32 bf16 = 512 x 16B chunks, 2 per thread
#pragma unroll
    for (int it = 0; it < 2; ++it) {
        int chunk = t + it * 256;             // 0..511
        int row   = chunk >> 2;
        int col8  = (chunk & 3) * 8;
        int grow  = blockM + row;
        if (grow > M - 1) grow = M - 1;       // clamp tail: no divergence, valid data
        async_cp16(lds_addr32(AsBuf + row * KPAD + col8),
                   A + (size_t)grow * K + k0 + col8);
    }
    // B tile: 64 n-rows x 32 bf16 = 256 x 16B chunks, 1 per thread
    {
        int row = t >> 2;                     // 0..63
        int c8  = (t & 3) * 8;
        async_cp16(lds_addr32(BsBuf + row * KPAD + c8),
                   Bt + (size_t)(blockN + row) * K + k0 + c8);
    }
}

template <bool RELU, bool OUTBF16>
__global__ __launch_bounds__(256) void gemm_bf16_kernel(
    const unsigned short* __restrict__ A, const unsigned short* __restrict__ Bt,
    const float* __restrict__ bias, void* __restrict__ Cout,
    int M, int K, int N)
{
    __shared__ __align__(16) unsigned short As[2][128 * KPAD];  // 2 x 10KB
    __shared__ __align__(16) unsigned short Bs[2][64 * KPAD];   // 2 x 5KB

    const int t      = threadIdx.x;
    const int lane   = t & 31;
    const int waveId = t >> 5;
    const int wm     = waveId & 3;      // wave M position (0..3) -> rows wm*32
    const int wn     = waveId >> 2;     // wave N position (0..1) -> cols wn*32
    const int half   = lane >> 4;       // K-half selector within fragment
    const int l16    = lane & 15;

    const int blockM = blockIdx.x * 128;
    const int blockN = blockIdx.y * 64;

    v8f acc[2][2] = {};

    // prologue: stage first tile into buffer 0
    stage_tile_async(A, Bt, As[0], Bs[0], blockM, blockN, M, K, 0, t);

    int cur = 0;
    for (int k0 = 0; k0 < K; k0 += 32) {
        if (k0 + 32 < K) {
            // prefetch next tile into the other buffer, then wait for current
            stage_tile_async(A, Bt, As[cur ^ 1], Bs[cur ^ 1],
                             blockM, blockN, M, K, k0 + 32, t);
            asm volatile("s_wait_asynccnt 3" ::: "memory");
        } else {
            asm volatile("s_wait_asynccnt 0" ::: "memory");
        }
        __syncthreads();

        // ---- build fragments (documented wave32 VGPR layouts)
        Frag a[2], b[2];
#pragma unroll
        for (int i = 0; i < 2; ++i) {
            int row = wm * 32 + i * 16 + l16;
            // lane holds K = {8h..8h+7} then {16+8h..16+8h+7}
            a[i].q[0] = *(const uint4*)(&As[cur][row * KPAD + 8 * half]);
            a[i].q[1] = *(const uint4*)(&As[cur][row * KPAD + 16 + 8 * half]);
        }
#pragma unroll
        for (int j = 0; j < 2; ++j) {
            int col = wn * 32 + j * 16 + l16;
            // lane holds K = {8h..8h+7} then {16+8h..16+8h+7} for its column
            b[j].q[0] = *(const uint4*)(&Bs[cur][col * KPAD + 8 * half]);
            b[j].q[1] = *(const uint4*)(&Bs[cur][col * KPAD + 16 + 8 * half]);
        }

#pragma unroll
        for (int i = 0; i < 2; ++i)
#pragma unroll
            for (int j = 0; j < 2; ++j)
                acc[i][j] = __builtin_amdgcn_wmma_f32_16x16x32_bf16(
                    false, a[i].v, false, b[j].v, (short)0, acc[i][j],
                    false, false);

        __syncthreads();
        cur ^= 1;
    }

    // ---- epilogue: C layout VGPR r, lane L -> M = r + 8*(L>>4), N = L&15
#pragma unroll
    for (int i = 0; i < 2; ++i) {
#pragma unroll
        for (int j = 0; j < 2; ++j) {
            int gcol = blockN + wn * 32 + j * 16 + l16;
            float bb = bias[gcol];
#pragma unroll
            for (int r = 0; r < 8; ++r) {
                int grow = blockM + wm * 32 + i * 16 + r + 8 * half;
                if (grow < M) {
                    float v = acc[i][j][r] + bb;
                    if (RELU) v = fmaxf(v, 0.0f);
                    if (OUTBF16)
                        ((unsigned short*)Cout)[(size_t)grow * N + gcol] = f2bf(v);
                    else
                        ((float*)Cout)[(size_t)grow * N + gcol] = v;
                }
            }
        }
    }
}

// ---------------------------------------------------------------------------
// Host-side orchestration
// ---------------------------------------------------------------------------
static inline size_t align_up(size_t x, size_t a) { return (x + a - 1) & ~(a - 1); }

extern "C" void kernel_launch(void* const* d_in, const int* in_sizes, int n_in,
                              void* d_out, int out_size, void* d_ws, size_t ws_size,
                              hipStream_t stream)
{
    (void)in_sizes; (void)n_in; (void)out_size; (void)ws_size;

    const float* feat    = (const float*)d_in[0];
    const float* Wself0  = (const float*)d_in[1];
    const float* Wneigh0 = (const float*)d_in[2];
    const float* b0      = (const float*)d_in[3];
    const float* Wself1  = (const float*)d_in[4];
    const float* Wneigh1 = (const float*)d_in[5];
    const float* b1      = (const float*)d_in[6];
    const int* es0 = (const int*)d_in[7];
    const int* ed0 = (const int*)d_in[8];
    const int* es1 = (const int*)d_in[9];
    const int* ed1 = (const int*)d_in[10];

    // workspace carve-out
    char* ws = (char*)d_ws;
    size_t off = 0;
    auto carve = [&](size_t bytes) { size_t o = off; off = align_up(off + bytes, 256); return (void*)(ws + o); };

    float*          msg0 = (float*)         carve((size_t)N_DST0 * D_IN  * 4);  // 40000x256 f32
    float*          deg0 = (float*)         carve((size_t)N_DST0        * 4);
    float*          msg1 = (float*)         carve((size_t)N_DST1 * D_HID * 4);  // 8000x512 f32
    float*          deg1 = (float*)         carve((size_t)N_DST1        * 4);
    unsigned short* X0   = (unsigned short*)carve((size_t)N_DST0 * 2 * D_IN  * 2); // 40000x512 bf16
    unsigned short* W0t  = (unsigned short*)carve((size_t)D_HID * (2 * D_IN) * 2); // 512x512 bf16 (N x K)
    unsigned short* H    = (unsigned short*)carve((size_t)N_DST0 * D_HID * 2);     // 40000x512 bf16
    unsigned short* X1   = (unsigned short*)carve((size_t)N_DST1 * 2 * D_HID * 2); // 8000x1024 bf16
    unsigned short* W1t  = (unsigned short*)carve((size_t)D_OUT * (2 * D_HID) * 2);// 256x1024 bf16 (N x K)

    // zero accumulators (graph-capture-safe async memsets)
    hipMemsetAsync(msg0, 0, (size_t)N_DST0 * D_IN  * 4, stream);
    hipMemsetAsync(deg0, 0, (size_t)N_DST0        * 4, stream);
    hipMemsetAsync(msg1, 0, (size_t)N_DST1 * D_HID * 4, stream);
    hipMemsetAsync(deg1, 0, (size_t)N_DST1        * 4, stream);

    // weight conversion / stacking / transpose (tiny)
    cvt_w0_kernel<<<(512 * 512 + 255) / 256, 256, 0, stream>>>(Wself0, Wneigh0, W0t);
    cvt_w1_kernel<<<(256 * 1024 + 255) / 256, 256, 0, stream>>>(Wself1, Wneigh1, W1t);

    // layer 0: aggregate, build concat input, fused GEMM (+bias, ReLU) -> H (bf16)
    scatter0_kernel<<<(E0 + 7) / 8, 256, 0, stream>>>(feat, es0, ed0, msg0, deg0, E0);
    build_x0_kernel<<<(N_DST0 * D_IN + 255) / 256, 256, 0, stream>>>(feat, msg0, deg0, X0);
    {
        dim3 grid((N_DST0 + 127) / 128, D_HID / 64);
        gemm_bf16_kernel<true, true><<<grid, 256, 0, stream>>>(
            X0, W0t, b0, (void*)H, N_DST0, 2 * D_IN, D_HID);
    }

    // layer 1: aggregate over H, build concat input, fused GEMM (+bias) -> d_out (f32)
    scatter1_kernel<<<(E1 + 7) / 8, 256, 0, stream>>>(H, es1, ed1, msg1, deg1, E1);
    build_x1_kernel<<<(N_DST1 * D_HID + 255) / 256, 256, 0, stream>>>(H, msg1, deg1, X1);
    {
        dim3 grid((N_DST1 + 127) / 128, D_OUT / 64);
        gemm_bf16_kernel<false, false><<<grid, 256, 0, stream>>>(
            X1, W1t, b1, d_out, N_DST1, 2 * D_HID, D_OUT);
    }
}
